// LiftingWaveletConv2D_4973572129292
// MI455X (gfx1250) — compile-verified, matrix operand order checked by reading, not run on hardware
//
#include <hip/hip_runtime.h>
#include <hip/hip_bf16.h>
#include <math.h>

// MI455X (gfx1250) implementation.
// Memory-bound (~5.4 GFLOP over ~0.6 GB HBM traffic -> ~27 us roofline at
// 23.3 TB/s). Channel matmuls run on v_wmma_f32_16x16x32_f16 (f16 A/B, f32
// accumulate); B fragments are register-resident across an M-tile loop so the
// LDS weight staging is amortized 8x per block.

typedef __attribute__((ext_vector_type(16))) _Float16 v16h;
typedef __attribute__((ext_vector_type(8)))  float    v8f;

#define C_CH      64
#define SB_ELEMS  (8*64*64*64)         // 2,097,152 per level-2 subband
#define LL1_ELEMS (8*128*128*64)       // 8,388,608
#define MT_PER_BLOCK 16                // M-tiles (of 16 pixels) per block

__device__ __forceinline__ float gelu_tanh(float x) {
    float x3 = x * x * x;
    return 0.5f * x * (1.0f + tanhf(0.7978845608028654f * (x + 0.044715f * x3)));
}

// ---------------------------------------------------------------------------
// WMMA fragment helpers. 16-bit A 16x32 layout: lane<16 -> M=lane, K in
// {kb+0..7, kb+16..23}, kb=(lane>=16)*8; B mirrors with N. C/D: lane -> N,
// VGPR r -> M (+8 for lanes 16..31).
// ---------------------------------------------------------------------------
__device__ __forceinline__ void load_B_frags(const _Float16* Wlds, int nTile, int lane,
                                             v16h& b0, v16h& b1) {
    const int kBase = (lane >> 4) << 3;
    const int colB  = (lane & 15) + nTile * 16;
    union { v16h v; _Float16 e[16]; } t0, t1;
#pragma unroll
    for (int j = 0; j < 16; ++j) {
        int k = kBase + (j & 7) + ((j >> 3) << 4);
        t0.e[j] = Wlds[k * C_CH + colB];
        t1.e[j] = Wlds[(k + 32) * C_CH + colB];
    }
    b0 = t0.v; b1 = t1.v;
}

__device__ __forceinline__ v8f wmma_tile_A(const float* __restrict__ act, int pixBase,
                                           v16h b0, v16h b1, int lane) {
    const int rowA  = lane & 15;
    const int kBase = (lane >> 4) << 3;
    union { v16h v; _Float16 e[16]; } a0, a1;
    const float* arow = act + (pixBase + rowA) * C_CH;
#pragma unroll
    for (int j = 0; j < 16; ++j) {
        int k = kBase + (j & 7) + ((j >> 3) << 4);
        a0.e[j] = (_Float16)arow[k];
        a1.e[j] = (_Float16)arow[k + 32];
    }
    v8f c = {};
    c = __builtin_amdgcn_wmma_f32_16x16x32_f16(false, a0.v, false, b0, (short)0, c, false, false);
    c = __builtin_amdgcn_wmma_f32_16x16x32_f16(false, a1.v, false, b1, (short)0, c, false, false);
    return c;
}

// ---------------------------------------------------------------------------
// K1: forward lifting level 1 -> ll1 only (level-1 details are never used).
// 1D low-pass: s[n] = -1/8 x[2n-2] + 1/4 x[2n-1] + 3/4 x[2n] + 1/4 x[2n+1] - 1/8 x[2n+2]
// ---------------------------------------------------------------------------
__global__ void fwd1_kernel(const float* __restrict__ x, float* __restrict__ ll1) {
    int tid = blockIdx.x * 256 + threadIdx.x;          // (b,h2,w2,c) flat
    int c  = tid & 63;
    int w2 = (tid >> 6) & 127;
    int h2 = (tid >> 13) & 127;
    int b  = tid >> 20;
    const float L[5] = {-0.125f, 0.25f, 0.75f, 0.25f, -0.125f};
    const float* xb = x + ((size_t)b << 22);           // 256*256*64
    float acc = 0.f;
#pragma unroll
    for (int i = 0; i < 5; ++i) {
        int hs = (2 * h2 - 2 + i + 256) & 255;
        float row = 0.f;
#pragma unroll
        for (int j = 0; j < 5; ++j) {
            int ws = (2 * w2 - 2 + j + 256) & 255;
            row = fmaf(L[j], xb[((hs << 8) + ws) * C_CH + c], row);
        }
        acc = fmaf(L[i], row, acc);
    }
    ll1[tid] = acc;
}

// ---------------------------------------------------------------------------
// K2: forward lifting level 2 -> ll2, lh2, hl2, hh2 from one 5x5 window.
// high-pass: d[n] = -1/2 x[2n] + x[2n+1] - 1/2 x[2n+2]  (window idx 2,3,4)
// ---------------------------------------------------------------------------
__global__ void fwd2_kernel(const float* __restrict__ ll1, float* __restrict__ s_all) {
    int tid = blockIdx.x * 256 + threadIdx.x;          // (b,h2,w2,c) flat @64x64
    int c  = tid & 63;
    int w2 = (tid >> 6) & 63;
    int h2 = (tid >> 12) & 63;
    int b  = tid >> 18;
    const float* xb = ll1 + ((size_t)b << 20);         // 128*128*64
    float lowW[5], highW[5];
#pragma unroll
    for (int i = 0; i < 5; ++i) {
        int hs = (2 * h2 - 2 + i + 128) & 127;
        float xv[5];
#pragma unroll
        for (int j = 0; j < 5; ++j) {
            int ws = (2 * w2 - 2 + j + 128) & 127;
            xv[j] = xb[((hs << 7) + ws) * C_CH + c];
        }
        lowW[i]  = -0.125f*xv[0] + 0.25f*xv[1] + 0.75f*xv[2] + 0.25f*xv[3] - 0.125f*xv[4];
        highW[i] = -0.5f*xv[2] + xv[3] - 0.5f*xv[4];
    }
    float ll = -0.125f*lowW[0] + 0.25f*lowW[1] + 0.75f*lowW[2] + 0.25f*lowW[3] - 0.125f*lowW[4];
    float lh = -0.5f*lowW[2] + lowW[3] - 0.5f*lowW[4];
    float hl = -0.125f*highW[0] + 0.25f*highW[1] + 0.75f*highW[2] + 0.25f*highW[3] - 0.125f*highW[4];
    float hh = -0.5f*highW[2] + highW[3] - 0.5f*highW[4];
    s_all[tid]               = ll;
    s_all[SB_ELEMS + tid]    = lh;
    s_all[2*SB_ELEMS + tid]  = hl;
    s_all[3*SB_ELEMS + tid]  = hh;
}

// ---------------------------------------------------------------------------
// K3: per-pixel 64x64 channel GEMM on each level-2 subband (WMMA).
// subband 0: w_approx[i*64+o]; subband s>=1: w_detail[(i*64+o)*3 + (s-1)]
// Block: 256 thr = 8 waves; B frags register-resident; 8 M-tile iters/wave.
// Grid: 4 subbands x 128 blocks (16 M-tiles each, 2048 M-tiles per subband).
// ---------------------------------------------------------------------------
__global__ void subband_gemm_kernel(const float* __restrict__ s_all,
                                    const float* __restrict__ w_approx,
                                    const float* __restrict__ w_detail,
                                    float* __restrict__ y_all) {
    __shared__ _Float16 Wlds[64 * 64];
    int t = threadIdx.x;
    int sb = blockIdx.x >> 7;
    int blkInSb = blockIdx.x & 127;
    for (int i = t; i < 64 * 64; i += 256) {
        float w = (sb == 0) ? w_approx[i] : w_detail[i * 3 + (sb - 1)];
        Wlds[i] = (_Float16)w;
    }
    __syncthreads();
    const float* act = s_all + (size_t)sb * SB_ELEMS;
    float* outp      = y_all + (size_t)sb * SB_ELEMS;
    int wave = t >> 5, lane = t & 31;
    int nTile = wave & 3;
    int mSub  = wave >> 2;
    v16h b0, b1;
    load_B_frags(Wlds, nTile, lane, b0, b1);
    int mOff = (lane >> 4) << 3;
    int n = (lane & 15) + nTile * 16;
#pragma unroll
    for (int it = 0; it < MT_PER_BLOCK / 2; ++it) {
        int mTile = blkInSb * MT_PER_BLOCK + it * 2 + mSub;
        int pixBase = mTile * 16;
        union { v8f v; float e[8]; } cc;
        cc.v = wmma_tile_A(act, pixBase, b0, b1, lane);
#pragma unroll
        for (int r = 0; r < 8; ++r)
            outp[(pixBase + r + mOff) * C_CH + n] = cc.e[r];
    }
}

// ---------------------------------------------------------------------------
// K4: inverse lifting level 2. Each thread rebuilds a 2x2 quad at 128x128 from
// the 3x3 neighborhood of the four (matmuled) subbands, periodic wrap.
// ---------------------------------------------------------------------------
__global__ void inv2_kernel(const float* __restrict__ y_all, float* __restrict__ c128) {
    int tid = blockIdx.x * 256 + threadIdx.x;          // (b,hn,wn,c) @64x64
    int c  = tid & 63;
    int wn = (tid >> 6) & 63;
    int hn = (tid >> 12) & 63;
    int b  = tid >> 18;
    const float* yll = y_all;
    const float* ylh = y_all + SB_ELEMS;
    const float* yhl = y_all + 2*SB_ELEMS;
    const float* yhh = y_all + 3*SB_ELEMS;
    int hm = (hn + 63) & 63;
    int hp = (hn + 1) & 63;
    auto idx = [&](int h_, int w_) { return ((((b << 6) + h_) << 6) + w_) * C_CH + c; };
    float lrE[3], lrO[3], hrE[3], hrO[3];
#pragma unroll
    for (int q = 0; q < 3; ++q) {
        int wq = (wn - 1 + q + 64) & 63;
        // l_row = inv(yll, ylh) along H
        float dh0 = ylh[idx(hn, wq)], dhm = ylh[idx(hm, wq)], dhp = ylh[idx(hp, wq)];
        float eL0 = yll[idx(hn, wq)] - 0.25f * (dh0 + dhm);
        float eL1 = yll[idx(hp, wq)] - 0.25f * (dhp + dh0);
        lrE[q] = eL0;
        lrO[q] = dh0 + 0.5f * (eL0 + eL1);
        // h_row = inv(yhl, yhh) along H
        float gh0 = yhh[idx(hn, wq)], ghm = yhh[idx(hm, wq)], ghp = yhh[idx(hp, wq)];
        float eH0 = yhl[idx(hn, wq)] - 0.25f * (gh0 + ghm);
        float eH1 = yhl[idx(hp, wq)] - 0.25f * (ghp + gh0);
        hrE[q] = eH0;
        hrO[q] = gh0 + 0.5f * (eH0 + eH1);
    }
    // inverse along W for output rows 2hn (E) and 2hn+1 (O)
    float ewE0 = lrE[1] - 0.25f * (hrE[1] + hrE[0]);
    float ewE1 = lrE[2] - 0.25f * (hrE[2] + hrE[1]);
    float ewO0 = lrO[1] - 0.25f * (hrO[1] + hrO[0]);
    float ewO1 = lrO[2] - 0.25f * (hrO[2] + hrO[1]);
    float oE = hrE[1] + 0.5f * (ewE0 + ewE1);
    float oO = hrO[1] + 0.5f * (ewO0 + ewO1);
    float* ob = c128 + ((size_t)b << 20);
    int h0 = hn << 1, w0 = wn << 1;
    ob[((h0 << 7) + w0) * C_CH + c]           = ewE0;
    ob[((h0 << 7) + w0 + 1) * C_CH + c]       = oE;
    ob[(((h0 + 1) << 7) + w0) * C_CH + c]     = ewO0;
    ob[(((h0 + 1) << 7) + w0 + 1) * C_CH + c] = oO;
}

// ---------------------------------------------------------------------------
// K5: fused residual WMMA GEMM (x @ dense_kernel + bias) + zero-detail level-1
// inverse (periodic 2x upsample of c128) + tanh-GELU -> out.
// 2048 blocks x 16 M-tiles; B frags register-resident per wave.
// ---------------------------------------------------------------------------
__global__ void final_kernel(const float* __restrict__ x,
                             const float* __restrict__ dense_kernel,
                             const float* __restrict__ dense_bias,
                             const float* __restrict__ c128,
                             float* __restrict__ out) {
    __shared__ _Float16 Wlds[64 * 64];
    int t = threadIdx.x;
    for (int i = t; i < 64 * 64; i += 256) Wlds[i] = (_Float16)dense_kernel[i];
    __syncthreads();
    int wave = t >> 5, lane = t & 31;
    int nTile = wave & 3;
    int mSub  = wave >> 2;
    v16h b0, b1;
    load_B_frags(Wlds, nTile, lane, b0, b1);
    int mOff = (lane >> 4) << 3;
    int n = (lane & 15) + nTile * 16;
    float bias = dense_bias[n];
#pragma unroll
    for (int it = 0; it < MT_PER_BLOCK / 2; ++it) {
        int mTile = blockIdx.x * MT_PER_BLOCK + it * 2 + mSub;
        int pixBase = mTile * 16;
        union { v8f v; float e[8]; } cc;
        cc.v = wmma_tile_A(x, pixBase, b0, b1, lane);
#pragma unroll
        for (int r = 0; r < 8; ++r) {
            int p = pixBase + r + mOff;                // flat (b,h,w)
            int b = p >> 16;
            int h = (p >> 8) & 255;
            int w = p & 255;
            int hn = h >> 1, wn = w >> 1;
            int hn1 = (hn + 1) & 127, wn1 = (wn + 1) & 127;
            const float* cb = c128 + ((size_t)b << 20); // 128*128*64
            float v00 = cb[((hn << 7) + wn) * C_CH + n];
            float wav;
            if (((h & 1) | (w & 1)) == 0) {
                wav = v00;
            } else if ((h & 1) == 0) {
                wav = 0.5f * (v00 + cb[((hn << 7) + wn1) * C_CH + n]);
            } else if ((w & 1) == 0) {
                wav = 0.5f * (v00 + cb[((hn1 << 7) + wn) * C_CH + n]);
            } else {
                wav = 0.25f * (v00 + cb[((hn << 7) + wn1) * C_CH + n]
                                   + cb[((hn1 << 7) + wn) * C_CH + n]
                                   + cb[((hn1 << 7) + wn1) * C_CH + n]);
            }
            out[(size_t)p * C_CH + n] = gelu_tanh(cc.e[r] + bias + wav);
        }
    }
}

extern "C" void kernel_launch(void* const* d_in, const int* in_sizes, int n_in,
                              void* d_out, int out_size, void* d_ws, size_t ws_size,
                              hipStream_t stream) {
    (void)in_sizes; (void)n_in; (void)out_size; (void)ws_size;
    const float* x            = (const float*)d_in[0];
    const float* w_approx     = (const float*)d_in[1];
    const float* w_detail     = (const float*)d_in[2];
    const float* dense_kernel = (const float*)d_in[3];
    const float* dense_bias   = (const float*)d_in[4];
    float* out = (float*)d_out;
    float* ws  = (float*)d_ws;

    float* ll1   = ws;                       // 8,388,608 f32 (reused as c128)
    float* s_all = ws + LL1_ELEMS;           // 4 x 2,097,152 f32
    float* y_all = ws + 2 * LL1_ELEMS;       // 4 x 2,097,152 f32
    float* c128  = ll1;                      // ll1 dead after fwd2

    fwd1_kernel        <<<LL1_ELEMS / 256, 256, 0, stream>>>(x, ll1);
    fwd2_kernel        <<<SB_ELEMS  / 256, 256, 0, stream>>>(ll1, s_all);
    // 2048 M-tiles per subband / 16 per block = 128 blocks per subband
    subband_gemm_kernel<<<4 * 128,         256, 0, stream>>>(s_all, w_approx, w_detail, y_all);
    inv2_kernel        <<<SB_ELEMS  / 256, 256, 0, stream>>>(y_all, c128);
    // 32768 M-tiles / 16 per block = 2048 blocks
    final_kernel       <<<2048,            256, 0, stream>>>(x, dense_kernel, dense_bias, c128, out);
}
